// AggressiveNet_87179246174848
// MI455X (gfx1250) — matrix operand order repaired and checked by reading, not compiled, not run on hardware
//
#include <hip/hip_runtime.h>

typedef __attribute__((ext_vector_type(16))) _Float16 v16h;
typedef __attribute__((ext_vector_type(8)))  float    v8f;
typedef __attribute__((ext_vector_type(2)))  _Float16 h2;
typedef int i4v __attribute__((vector_size(16)));

#if __has_builtin(__builtin_amdgcn_global_load_async_to_lds_b128)
#define ASYNC_OK 1
#endif

// ---------------------------------------------------------------------------
// Async global->LDS bulk copy (CDNA5 ASYNCcnt path), 16B granules.
// Builtin signature (probed via diagnostics): (int4 as1*, int4 as3*, imm, imm)
// Fallback: plain vector ld/st (still correct, just synchronous).
// ---------------------------------------------------------------------------
__device__ __forceinline__ void async_copy_to_lds(const void* g, void* l,
                                                  int bytes, int tid) {
#ifdef ASYNC_OK
  auto gp = (__attribute__((address_space(1))) i4v*)(unsigned long long)g;
  auto lp = (__attribute__((address_space(3))) i4v*)(unsigned long long)l;
  const int n16 = bytes >> 4;
  for (int i = tid; i < n16; i += 256) {
    __builtin_amdgcn_global_load_async_to_lds_b128(gp + i, lp + i, 0, 0);
  }
#else
  for (int off = tid * 16; off < bytes; off += 256 * 16) {
    *(uint4*)((char*)l + off) = *(const uint4*)((const char*)g + off);
  }
#endif
}

__device__ __forceinline__ void wait_async() {
#ifdef ASYNC_OK
#if __has_builtin(__builtin_amdgcn_s_wait_asynccnt)
  __builtin_amdgcn_s_wait_asynccnt(0);
#else
  asm volatile("s_wait_asynccnt 0" ::: "memory");
#endif
#endif
}

// ---------------------------------------------------------------------------
// Wave-level 16x16 WMMA tile helpers (f16 in, f32 accumulate)
// A: row-major [M][K] f16 in LDS; Bt: row-major [N][K] f16 in LDS.
// ---------------------------------------------------------------------------
__device__ __forceinline__ v8f wmma_tiles(const _Float16* A, int strideA,
                                          const _Float16* Bt, int strideB,
                                          int nt, int K, v8f acc) {
  const int lane = threadIdx.x & 31;
  const int kA = (lane & 16) ? 8 : 0;
  const int kB = (lane & 16) ? 8 : 0;
  const _Float16* ap0 = A + (lane & 15) * strideA + kA;
  const _Float16* bp0 = Bt + (nt * 16 + (lane & 15)) * strideB + kB;
  for (int k0 = 0; k0 < K; k0 += 32) {
    v16h a, b;
#pragma unroll
    for (int v = 0; v < 8; ++v) {
      const int kka = ((v >> 2) << 4) + ((v & 3) << 1);   // ISA 16-bit A striping
      h2 av = *(const h2*)(ap0 + k0 + kka);
      h2 bv = *(const h2*)(bp0 + k0 + kka);
      a[2 * v] = av.x; a[2 * v + 1] = av.y;
      b[2 * v] = bv.x; b[2 * v + 1] = bv.y;
    }
    acc = __builtin_amdgcn_wmma_f32_16x16x32_f16(false, a, false, b,
                                                 (short)0, acc, false, false);
  }
  return acc;
}

__device__ __forceinline__ v8f bias_acc(const float* bias, int nt) {
  float bv = bias[nt * 16 + (threadIdx.x & 15)];
  v8f acc;
#pragma unroll
  for (int r = 0; r < 8; ++r) acc[r] = bv;
  return acc;
}

__device__ __forceinline__ void store_tile_f32(float* Df, int strideD,
                                               int mt, int nt, const v8f& acc) {
  const int lane = threadIdx.x & 31;
  const int mbase = mt * 16 + ((lane & 16) ? 8 : 0);
  const int n = nt * 16 + (lane & 15);
#pragma unroll
  for (int r = 0; r < 8; ++r) Df[(mbase + r) * strideD + n] = acc[r];
}

__device__ __forceinline__ void store_tile_f16(_Float16* D, int strideD, int doff,
                                               int nt, const v8f& acc, bool act) {
  const int lane = threadIdx.x & 31;
  const int mbase = (lane & 16) ? 8 : 0;
  const int n = nt * 16 + (lane & 15);
#pragma unroll
  for (int r = 0; r < 8; ++r) {
    float v = acc[r];
    if (act) v = (v > 0.f) ? v : 0.01f * v;
    D[(mbase + r) * strideD + doff + n] = (_Float16)v;
  }
}

// ===========================================================================
// Weight prep: fp32 [K?,Cout] (optionally conv [2,CinReal,Cout]) ->
// f16 Bt blob [CoutRows][stride], transposed + zero-padded, matching the LDS
// layout consumed by wmma_tiles. Runs once; main kernels async-DMA the blobs.
// ===========================================================================
__global__ __launch_bounds__(256) void prep_wt_kernel(
    const float* __restrict__ W, _Float16* __restrict__ dst,
    int K, int CkIn, int CinReal, int CoutRows, int CoutReal, int stride) {
  const int total = CoutRows * stride;
  for (int idx = blockIdx.x * 256 + threadIdx.x; idx < total;
       idx += gridDim.x * 256) {
    int co = idx / stride, k = idx - co * stride;
    float v = 0.f;
    if (k < K && co < CoutReal) {
      int dt = (k >= CkIn) ? 1 : 0;
      int ci = k - dt * CkIn;
      if (ci < CinReal) v = W[(dt * CinReal + ci) * CoutReal + co];
    }
    dst[idx] = (_Float16)v;
  }
}

// ===========================================================================
// Kernel 1: fused pointnet. One (b,s) per workgroup, 8 waves. Double-buffered
// async weight staging overlaps the next layer's DMA with current WMMA.
// ===========================================================================
#define SA_STR 130
#define SF_STR 129

__device__ __forceinline__ void inorm_lrelu(const float* sF, _Float16* sA,
                                            const float* g, const float* b,
                                            float* sScale, float* sShift,
                                            float* sRedS, float* sRedQ,
                                            int Cout, int logC, int tid) {
  __syncthreads();
  // all 256 threads make partial sums: 256/Cout segments of 64/(256/Cout) rows
  const int c = tid & (Cout - 1);
  const int tpc = 256 >> logC;
  const int rows = 64 / tpc;
  const int r0 = (tid >> logC) * rows;
  float s = 0.f, sq = 0.f;
  for (int r = r0; r < r0 + rows; ++r) {
    float v = sF[r * SF_STR + c];
    s += v; sq += v * v;
  }
  sRedS[tid] = s; sRedQ[tid] = sq;
  __syncthreads();
  if (tid < Cout) {
    s = 0.f; sq = 0.f;
    for (int j = 0; j < tpc; ++j) {
      s += sRedS[tid + (j << logC)];
      sq += sRedQ[tid + (j << logC)];
    }
    float mu = s * (1.f / 64.f);
    float var = sq * (1.f / 64.f) - mu * mu;
    float rstd = rsqrtf(var + 1e-5f);
    float sc = rstd * g[tid];
    sScale[tid] = sc;
    sShift[tid] = b[tid] - mu * sc;
  }
  __syncthreads();
  for (int idx = tid; idx < 64 * Cout; idx += 256) {
    int r = idx >> logC, cc = idx & (Cout - 1);
    float v = sF[r * SF_STR + cc] * sScale[cc] + sShift[cc];
    v = (v > 0.f) ? v : 0.01f * v;             // LeakyReLU fused
    sA[r * SA_STR + cc] = (_Float16)v;
  }
  __syncthreads();
}

__device__ __forceinline__ void gemm64(const _Float16* sA, const _Float16* sW,
                                       float* sF, const float* bias,
                                       int K, int Nc, int tid) {
  const int wave = tid >> 5;
  const int numN = Nc >> 4;
  const int jobs = numN << 2;                  // 4 M-tiles x numN N-tiles
  for (int job = wave; job < jobs; job += 8) {
    int mt = job & 3, nt = job >> 2;
    v8f acc = bias_acc(bias, nt);
    acc = wmma_tiles(sA + mt * 16 * SA_STR, SA_STR, sW, K + 2, nt, K, acc);
    store_tile_f32(sF, SF_STR, mt, nt, acc);
  }
  __syncthreads();
}

__global__ __launch_bounds__(256) void pointnet_kernel(
    const float* __restrict__ fts,
    const float* __restrict__ pw1, const float* __restrict__ pb1,
    const float* __restrict__ pg1, const float* __restrict__ pbe1,
    const _Float16* __restrict__ wt2, const float* __restrict__ pb2,
    const float* __restrict__ pg2, const float* __restrict__ pbe2,
    const _Float16* __restrict__ wt3, const float* __restrict__ pb3,
    const float* __restrict__ pg3, const float* __restrict__ pbe3,
    const _Float16* __restrict__ wt4, const float* __restrict__ pb4,
    float* __restrict__ emb) {
  __shared__ __align__(16) _Float16 sA[64 * SA_STR];
  __shared__ __align__(16) _Float16 sW0[128 * 130];
  __shared__ __align__(16) _Float16 sW1[128 * 66];
  __shared__ float sF[64 * SF_STR];
  __shared__ float sScale[128], sShift[128];
  __shared__ float sRedS[256], sRedQ[256];

  const int tid = threadIdx.x;
  const int bs = blockIdx.x;                   // b*8 + s
  const float* x0 = fts + (size_t)bs * (64 * 5);

  // kick off layer-2 weight DMA immediately
  async_copy_to_lds(wt2, sW0, 64 * 34 * 2, tid);

  // layer 1: [64,5] @ [5,32] -- K=5, VALU
  for (int idx = tid; idx < 64 * 32; idx += 256) {
    const int n = idx >> 5, co = idx & 31;
    float acc = pb1[co];
#pragma unroll
    for (int c = 0; c < 5; ++c) acc += x0[n * 5 + c] * pw1[c * 32 + co];
    sF[n * SF_STR + co] = acc;
  }
  inorm_lrelu(sF, sA, pg1, pbe1, sScale, sShift, sRedS, sRedQ, 32, 5, tid);
  wait_async(); __syncthreads();               // wt2 resident

  async_copy_to_lds(wt3, sW1, 128 * 66 * 2, tid);   // overlap with layer 2
  gemm64(sA, sW0, sF, pb2, 32, 64, tid);
  inorm_lrelu(sF, sA, pg2, pbe2, sScale, sShift, sRedS, sRedQ, 64, 6, tid);
  wait_async(); __syncthreads();               // wt3 resident

  async_copy_to_lds(wt4, sW0, 128 * 130 * 2, tid);  // overlap with layer 3
  gemm64(sA, sW1, sF, pb3, 64, 128, tid);
  inorm_lrelu(sF, sA, pg3, pbe3, sScale, sShift, sRedS, sRedQ, 128, 7, tid);
  wait_async(); __syncthreads();               // wt4 resident

  gemm64(sA, sW0, sF, pb4, 128, 128, tid);     // layer 4 (bias only)

  // mean over the 64 points, parallel 2-stage reduction
  {
    const int c = tid & 127, seg = tid >> 7;
    float s = 0.f;
    for (int r = seg * 32; r < seg * 32 + 32; ++r) s += sF[r * SF_STR + c];
    sRedS[tid] = s;
  }
  __syncthreads();
  if (tid < 128)
    emb[(size_t)bs * 128 + tid] = (sRedS[tid] + sRedS[tid + 128]) * (1.f / 64.f);
}

// ===========================================================================
// Kernel 2: Conv1D stacks + denses + control head. 16 batch rows / block.
// Activations time-major f16: buf[b'][t*Ck + c], stride 1154; t=8 slot zeroed
// => k=2 'same' conv == GEMM with K = 2*Ck contiguous. Weights async-DMA'd.
// ===========================================================================
#define SX_STR 1154

__device__ __forceinline__ void conv_k2(const _Float16* X, int CkIn,
                                        _Float16* Y, int CkOut,
                                        const _Float16* wblob, int wbytes,
                                        const float* bias, _Float16* sW,
                                        bool act, int tid) {
  const int strideW = 2 * CkIn + 2;
  async_copy_to_lds(wblob, sW, wbytes, tid);
  for (int idx = tid; idx < 16 * CkOut; idx += 256) {  // zero t=8 pad of dest
    int b2 = idx / CkOut, c = idx - b2 * CkOut;
    Y[b2 * SX_STR + 8 * CkOut + c] = (_Float16)0.f;
  }
  wait_async();
  __syncthreads();
  const int wave = tid >> 5;
  const int numN = CkOut >> 4;
  const int jobs = 8 * numN;                   // t in [0,8) x N-tiles
  for (int job = wave; job < jobs; job += 8) {
    int t = job / numN, nt = job - (job / numN) * numN;
    v8f acc = bias_acc(bias, nt);
    acc = wmma_tiles(X + t * CkIn, SX_STR, sW, strideW, nt, 2 * CkIn, acc);
    store_tile_f16(Y + t * CkOut, SX_STR, 0, nt, acc, act);
  }
  __syncthreads();
}

__device__ __forceinline__ void dense512(const _Float16* X,
                                         const _Float16* wblob0,
                                         const _Float16* wblob1,
                                         const float* bias, _Float16* sE,
                                         int eoff, _Float16* sW, int tid) {
  const int nt = tid >> 5;                     // 8 N-tiles, one per wave
  v8f acc = bias_acc(bias, nt);
#pragma unroll
  for (int chunk = 0; chunk < 2; ++chunk) {    // K=512 in 2 chunks of 256
    __syncthreads();
    async_copy_to_lds(chunk ? wblob1 : wblob0, sW, 128 * 258 * 2, tid);
    wait_async();
    __syncthreads();
    acc = wmma_tiles(X + chunk * 256, SX_STR, sW, 258, nt, 256, acc);
  }
  store_tile_f16(sE, 258, eoff, nt, acc, false);
  __syncthreads();
}

__device__ __forceinline__ void head_gemm(const _Float16* A, int strideA,
                                          const _Float16* wblob, int wbytes,
                                          const float* bias, int K, int Nc,
                                          _Float16* D, int strideD, bool act,
                                          _Float16* sW, int tid) {
  __syncthreads();
  async_copy_to_lds(wblob, sW, wbytes, tid);
  wait_async();
  __syncthreads();
  const int wave = tid >> 5;
  if (wave < (Nc >> 4)) {
    v8f acc = bias_acc(bias, wave);
    acc = wmma_tiles(A, strideA, sW, K + 2, wave, K, acc);
    store_tile_f16(D, strideD, 0, wave, acc, act);
  }
  __syncthreads();
}

__global__ __launch_bounds__(256) void head_kernel(
    const float* __restrict__ emb, const float* __restrict__ state,
    const _Float16* __restrict__ bm1, const _Float16* __restrict__ bm2,
    const _Float16* __restrict__ bm3, const _Float16* __restrict__ bm4,
    const _Float16* __restrict__ bmd0, const _Float16* __restrict__ bmd1,
    const _Float16* __restrict__ bs1, const _Float16* __restrict__ bs2,
    const _Float16* __restrict__ bs3, const _Float16* __restrict__ bs4,
    const _Float16* __restrict__ bsd0, const _Float16* __restrict__ bsd1,
    const _Float16* __restrict__ bc1, const _Float16* __restrict__ bc2,
    const _Float16* __restrict__ bc3, const _Float16* __restrict__ bc4,
    const float* __restrict__ mb1, const float* __restrict__ mb2,
    const float* __restrict__ mb3, const float* __restrict__ mb4,
    const float* __restrict__ mdb,
    const float* __restrict__ sb1, const float* __restrict__ sb2,
    const float* __restrict__ sb3, const float* __restrict__ sb4,
    const float* __restrict__ sdb,
    const float* __restrict__ cb1, const float* __restrict__ cb2,
    const float* __restrict__ cb3, const float* __restrict__ cb4,
    float* __restrict__ out) {
  __shared__ __align__(16) _Float16 sX[16 * SX_STR];
  __shared__ __align__(16) _Float16 sY[16 * SX_STR];
  __shared__ __align__(16) _Float16 sW[128 * 258];
  __shared__ __align__(16) _Float16 sE[16 * 258];
  __shared__ __align__(16) _Float16 sH1[16 * 130];
  __shared__ __align__(16) _Float16 sH2[16 * 66];
  __shared__ __align__(16) _Float16 sH3[16 * 34];
  __shared__ float sO[16 * 16];

  const int tid = threadIdx.x;
  const int b0 = blockIdx.x * 16;

  // warm the embedding tile (global_prefetch_b8)
  const char* embBase = (const char*)(emb + (size_t)b0 * 8 * 128);
  for (int off = tid * 64; off < 16 * 8 * 128 * 4; off += 256 * 64)
    __builtin_prefetch(embBase + off, 0, 0);

  // ---- mergenet branch: emb [16][8][128] -> sX (Ck=128, t=8 zeroed) ----
  for (int idx = tid; idx < 16 * 8 * 128; idx += 256) {
    int b2 = idx >> 10, rem = idx & 1023;
    int t = rem >> 7, c = rem & 127;
    sX[b2 * SX_STR + (t << 7) + c] =
        (_Float16)emb[(((size_t)(b0 + b2)) * 8 + t) * 128 + c];
  }
  for (int idx = tid; idx < 16 * 128; idx += 256) {
    int b2 = idx >> 7, c = idx & 127;
    sX[b2 * SX_STR + 1024 + c] = (_Float16)0.f;
  }
  conv_k2(sX, 128, sY, 128, bm1, 128 * 258 * 2, mb1, sW, true, tid);
  conv_k2(sY, 128, sX, 64,  bm2, 64 * 258 * 2,  mb2, sW, true, tid);
  conv_k2(sX, 64,  sY, 64,  bm3, 64 * 130 * 2,  mb3, sW, true, tid);
  conv_k2(sY, 64,  sX, 64,  bm4, 64 * 130 * 2,  mb4, sW, true, tid);
  dense512(sX, bmd0, bmd1, mdb, sE, 0, sW, tid);   // fts_emb -> sE[:,0:128]

  // ---- states branch: state [16][8][36] -> sY (Ck padded to 48) ----
  for (int idx = tid; idx < 16 * 9 * 48; idx += 256) {
    int b2 = idx / 432, rem = idx - b2 * 432;
    int t = rem / 48, c = rem - t * 48;
    float v = (t < 8 && c < 36)
                  ? state[(((size_t)(b0 + b2)) * 8 + t) * 36 + c] : 0.f;
    sY[b2 * SX_STR + t * 48 + c] = (_Float16)v;
  }
  conv_k2(sY, 48,  sX, 128, bs1, 128 * 98 * 2,  sb1, sW, true,  tid);
  conv_k2(sX, 128, sY, 64,  bs2, 64 * 258 * 2,  sb2, sW, true,  tid);
  conv_k2(sY, 64,  sX, 64,  bs3, 64 * 130 * 2,  sb3, sW, true,  tid);
  conv_k2(sX, 64,  sY, 64,  bs4, 64 * 130 * 2,  sb4, sW, false, tid);
  dense512(sY, bsd0, bsd1, sdb, sE, 128, sW, tid); // st_emb -> sE[:,128:256]

  // ---- control head ----
  head_gemm(sE, 258, bc1, 128 * 258 * 2, cb1, 256, 128, sH1, 130, true, sW, tid);
  head_gemm(sH1, 130, bc2, 64 * 130 * 2, cb2, 128, 64, sH2, 66, true, sW, tid);
  head_gemm(sH2, 66, bc3, 32 * 66 * 2, cb3, 64, 32, sH3, 34, true, sW, tid);

  // final [16,32] @ [32,4] (N padded to 16)
  __syncthreads();
  async_copy_to_lds(bc4, sW, 16 * 34 * 2, tid);
  wait_async();
  __syncthreads();
  if ((tid >> 5) == 0) {
    const int n = tid & 15;
    float bv = (n < 4) ? cb4[n] : 0.f;
    v8f acc;
#pragma unroll
    for (int r = 0; r < 8; ++r) acc[r] = bv;
    acc = wmma_tiles(sH3, 34, sW, 34, 0, 32, acc);
    store_tile_f32(sO, 16, 0, 0, acc);
  }
  __syncthreads();
  if (tid < 64) {
    int b2 = tid >> 2, j = tid & 3;
    float v = sO[b2 * 16 + j];
    float r = (j == 0) ? 21.f / (1.f + expf(-v)) : 6.f * tanhf(v);
    out[((size_t)(b0 + b2)) * 4 + j] = r;
  }
}

// ===========================================================================
extern "C" void kernel_launch(void* const* d_in, const int* in_sizes, int n_in,
                              void* d_out, int out_size, void* d_ws, size_t ws_size,
                              hipStream_t stream) {
  (void)in_sizes; (void)n_in; (void)out_size; (void)ws_size;
  const float* fts   = (const float*)d_in[0];
  const float* state = (const float*)d_in[1];
  const float *pw1 = (const float*)d_in[2],  *pb1 = (const float*)d_in[3];
  const float *pg1 = (const float*)d_in[4],  *pbe1 = (const float*)d_in[5];
  const float *pw2 = (const float*)d_in[6],  *pb2 = (const float*)d_in[7];
  const float *pg2 = (const float*)d_in[8],  *pbe2 = (const float*)d_in[9];
  const float *pw3 = (const float*)d_in[10], *pb3 = (const float*)d_in[11];
  const float *pg3 = (const float*)d_in[12], *pbe3 = (const float*)d_in[13];
  const float *pw4 = (const float*)d_in[14], *pb4 = (const float*)d_in[15];
  const float *mw1 = (const float*)d_in[16], *mb1 = (const float*)d_in[17];
  const float *mw2 = (const float*)d_in[18], *mb2 = (const float*)d_in[19];
  const float *mw3 = (const float*)d_in[20], *mb3 = (const float*)d_in[21];
  const float *mw4 = (const float*)d_in[22], *mb4 = (const float*)d_in[23];
  const float *mdw = (const float*)d_in[24], *mdb = (const float*)d_in[25];
  const float *sw1 = (const float*)d_in[26], *sb1 = (const float*)d_in[27];
  const float *sw2 = (const float*)d_in[28], *sb2 = (const float*)d_in[29];
  const float *sw3 = (const float*)d_in[30], *sb3 = (const float*)d_in[31];
  const float *sw4 = (const float*)d_in[32], *sb4 = (const float*)d_in[33];
  const float *sdw = (const float*)d_in[34], *sdb = (const float*)d_in[35];
  const float *cw1 = (const float*)d_in[36], *cb1 = (const float*)d_in[37];
  const float *cw2 = (const float*)d_in[38], *cb2 = (const float*)d_in[39];
  const float *cw3 = (const float*)d_in[40], *cb3 = (const float*)d_in[41];
  const float *cw4 = (const float*)d_in[42], *cb4 = (const float*)d_in[43];

  // ---- workspace layout ----
  char* ws = (char*)d_ws;
  float* emb = (float*)ws;                                 // [8192,128] f32
  char* p = ws + (size_t)8192 * 128 * 4;
  auto alloc = [&](size_t bytes) { _Float16* r = (_Float16*)p; p += bytes; return r; };
  _Float16* wt2  = alloc(64 * 34 * 2);
  _Float16* wt3  = alloc(128 * 66 * 2);
  _Float16* wt4  = alloc(128 * 130 * 2);
  _Float16* bm1  = alloc(128 * 258 * 2);
  _Float16* bm2  = alloc(64 * 258 * 2);
  _Float16* bm3  = alloc(64 * 130 * 2);
  _Float16* bm4  = alloc(64 * 130 * 2);
  _Float16* bmd0 = alloc(128 * 258 * 2);
  _Float16* bmd1 = alloc(128 * 258 * 2);
  _Float16* bs1  = alloc(128 * 98 * 2);
  _Float16* bs2  = alloc(64 * 258 * 2);
  _Float16* bs3  = alloc(64 * 130 * 2);
  _Float16* bs4  = alloc(64 * 130 * 2);
  _Float16* bsd0 = alloc(128 * 258 * 2);
  _Float16* bsd1 = alloc(128 * 258 * 2);
  _Float16* bc1  = alloc(128 * 258 * 2);
  _Float16* bc2  = alloc(64 * 130 * 2);
  _Float16* bc3  = alloc(32 * 66 * 2);
  _Float16* bc4  = alloc(16 * 34 * 2);

  // ---- weight prep (transpose + f16 + pad, once) ----
  auto PREP = [&](const float* W, _Float16* dst, int K, int CkIn, int CinReal,
                  int CoutRows, int CoutReal, int stride) {
    prep_wt_kernel<<<32, 256, 0, stream>>>(W, dst, K, CkIn, CinReal,
                                           CoutRows, CoutReal, stride);
  };
  PREP(pw2, wt2, 32, 32, 32, 64, 64, 34);
  PREP(pw3, wt3, 64, 64, 64, 128, 128, 66);
  PREP(pw4, wt4, 128, 128, 128, 128, 128, 130);
  PREP(mw1, bm1, 256, 128, 128, 128, 128, 258);
  PREP(mw2, bm2, 256, 128, 128, 64, 64, 258);
  PREP(mw3, bm3, 128, 64, 64, 64, 64, 130);
  PREP(mw4, bm4, 128, 64, 64, 64, 64, 130);
  PREP(mdw, bmd0, 256, 256, 256, 128, 128, 258);
  PREP(mdw + 256 * 128, bmd1, 256, 256, 256, 128, 128, 258);
  PREP(sw1, bs1, 96, 48, 36, 128, 128, 98);
  PREP(sw2, bs2, 256, 128, 128, 64, 64, 258);
  PREP(sw3, bs3, 128, 64, 64, 64, 64, 130);
  PREP(sw4, bs4, 128, 64, 64, 64, 64, 130);
  PREP(sdw, bsd0, 256, 256, 256, 128, 128, 258);
  PREP(sdw + 256 * 128, bsd1, 256, 256, 256, 128, 128, 258);
  PREP(cw1, bc1, 256, 256, 256, 128, 128, 258);
  PREP(cw2, bc2, 128, 128, 128, 64, 64, 130);
  PREP(cw3, bc3, 64, 64, 64, 32, 32, 66);
  PREP(cw4, bc4, 32, 32, 32, 16, 4, 34);

  pointnet_kernel<<<1024 * 8, 256, 0, stream>>>(
      fts, pw1, pb1, pg1, pbe1, wt2, pb2, pg2, pbe2,
      wt3, pb3, pg3, pbe3, wt4, pb4, emb);

  head_kernel<<<64, 256, 0, stream>>>(
      emb, state,
      bm1, bm2, bm3, bm4, bmd0, bmd1,
      bs1, bs2, bs3, bs4, bsd0, bsd1,
      bc1, bc2, bc3, bc4,
      mb1, mb2, mb3, mb4, mdb,
      sb1, sb2, sb3, sb4, sdb,
      cb1, cb2, cb3, cb4,
      (float*)d_out);
}
